// RNNModel_20409684590955
// MI455X (gfx1250) — matrix operand order repaired and checked by reading, not compiled, not run on hardware
//
#include <hip/hip_runtime.h>
#include <hip/hip_bf16.h>

#define VOCAB 8192
#define HID   256
#define BATCH 64
#define SEQ   256
#define LDK   264   // padded LDS row stride in bf16 elements (528B row: 16B aligned, bank-conflict free)

typedef __attribute__((ext_vector_type(16))) __bf16 v16bf;
typedef __attribute__((ext_vector_type(8)))  float  v8f;
typedef unsigned int u32x4 __attribute__((ext_vector_type(4)));
typedef int          i32x8 __attribute__((ext_vector_type(8)));
typedef int          i32x4 __attribute__((ext_vector_type(4)));

union FragAB { unsigned int u[8]; v16bf v; };
union FragC  { v8f v; float f[8]; };

__device__ __forceinline__ unsigned int cvt_bf16(float x) {
    unsigned int u = __float_as_uint(x);
    u += 0x7FFFu + ((u >> 16) & 1u);   // round-to-nearest-even
    return u >> 16;
}
__device__ __forceinline__ unsigned int pack_bf16(float lo, float hi) {
    return cvt_bf16(lo) | (cvt_bf16(hi) << 16);
}

// ---------------------------------------------------------------------------
// TDM: load a [rows x 256] bf16 tile (row stride 256 elements in memory) into
// LDS with +16B padding per 512B row (i.e. LDS row stride = LDK bf16 elems).
// Issue from ONE wave only; follow with s_wait_tensorcnt + barrier.
// ---------------------------------------------------------------------------
__device__ __forceinline__ void tdm_load_tile_bf16(
    unsigned int lds_addr, const unsigned short* gptr,
    unsigned int tile_rows, unsigned int tensor_rows)
{
    unsigned long long ga = (unsigned long long)(uintptr_t)gptr;
    u32x4 g0;
    g0[0] = 1u;                                       // count=1, user desc, no gather
    g0[1] = lds_addr;                                 // lds_addr [63:32]
    g0[2] = (unsigned int)(ga & 0xFFFFFFFFu);         // global_addr[31:0]
    g0[3] = (unsigned int)((ga >> 32) & 0x01FFFFFFu)  // global_addr[56:32]
          | (2u << 30);                               // type=2 ("image")
    i32x8 g1;
    g1[0] = (int)((1u << 16)        // data_size = 2 bytes
                | (1u << 20)        // pad_enable
                | (6u << 22)        // pad_interval: 128 DWORDs = 512B
                | (3u << 25));      // pad_amount:  4 DWORDs = 16B
    g1[1] = (int)(256u << 16);      // tensor_dim0[15:0]=256 at bits 63:48
    g1[2] = (int)((tensor_rows & 0xFFFFu) << 16);     // tensor_dim1[15:0] at bits 95:80
    g1[3] = (int)(((tensor_rows >> 16) & 0xFFFFu)     // tensor_dim1[31:16]
                | (256u << 16));                      // tile_dim0 = 256
    g1[4] = (int)(tile_rows & 0xFFFFu);               // tile_dim1; tile_dim2=0
    g1[5] = (int)256;                                 // tensor_dim0_stride[31:0] = 256
    g1[6] = 0;                                        // stride hi, dim1_stride lo
    g1[7] = 0;
    i32x4 z4 = {0, 0, 0, 0};
    i32x8 z8 = {0, 0, 0, 0, 0, 0, 0, 0};
    __builtin_amdgcn_tensor_load_to_lds(g0, g1, z4, z4, z8, 0);
}

// A fragment (16x32 bf16, M = lane&15). vgpr v holds k-pair at
// 2v + 16*(lane>=16), packed {k,[15:0]},{k+1,[31:16]}.
__device__ __forceinline__ v16bf load_frag_a(const unsigned short* base, int row, int ks, int lane) {
    FragAB f;
    const unsigned short* p = base + row * LDK + ks + ((lane >> 4) << 4);
#pragma unroll
    for (int v = 0; v < 8; ++v) {
        int k = 2 * v;
        f.u[v] = *(const unsigned int*)(p + k);
    }
    return f.v;
}

// B fragment (32x16 bf16, N = lane&15). vgpr v holds k-pair 2v + 16*(lane>=16).
// base is row-major [n][k], i.e. base holds B^T.
__device__ __forceinline__ v16bf load_frag_b(const unsigned short* base, int col, int ks, int lane) {
    FragAB f;
    const unsigned short* p = base + col * LDK + ks + ((lane >> 4) << 4);
#pragma unroll
    for (int v = 0; v < 8; ++v)
        f.u[v] = *(const unsigned int*)(p + 2 * v);
    return f.v;
}

// ---------------------------------------------------------------------------
// Stage 1: one-time f32 -> bf16 conversion of W_out and W_hh into workspace,
// so the GEMM stages can TDM-copy pure bf16 tiles.
// ---------------------------------------------------------------------------
#define WOUT_CHUNKS ((VOCAB * HID) / 8)          // 262144
#define WHH_CHUNKS  ((HID * HID) / 8)            // 8192
__global__ __launch_bounds__(256) void convert_weights_kernel(
    const float* __restrict__ W_out, const float* __restrict__ W_hh,
    unsigned short* __restrict__ Wout_bf, unsigned short* __restrict__ Whh_bf)
{
    int i = blockIdx.x * 256 + threadIdx.x;      // chunk of 8 floats
    const float* src;
    unsigned short* dst;
    int off;
    if (i < WOUT_CHUNKS) { src = W_out; dst = Wout_bf; off = i * 8; }
    else if (i < WOUT_CHUNKS + WHH_CHUNKS) { src = W_hh; dst = Whh_bf; off = (i - WOUT_CHUNKS) * 8; }
    else return;
    const float4* s = (const float4*)(src + off);
    float4 f0 = s[0], f1 = s[1];
    uint4 pk = { pack_bf16(f0.x, f0.y), pack_bf16(f0.z, f0.w),
                 pack_bf16(f1.x, f1.y), pack_bf16(f1.z, f1.w) };
    *(uint4*)(dst + off) = pk;
}

// ---------------------------------------------------------------------------
// Stage 2: recurrent scan. 4 blocks x 256 threads; block owns 16 batch rows.
// h_new = tanh( gather(W_ih, X[:,t]) + b_ih + h @ W_hh^T + b_hh )
// ---------------------------------------------------------------------------
__global__ __launch_bounds__(256) void rnn_scan_kernel(
    const int*   __restrict__ X,       // (B, T) int32
    const float* __restrict__ state,   // (1, B, H)
    const float* __restrict__ W_ih,    // (H, V)
    const unsigned short* __restrict__ Whh_bf,  // (H, H) bf16
    const float* __restrict__ b_ih,    // (H)
    const float* __restrict__ b_hh,    // (H)
    unsigned short* __restrict__ Ybf,  // (T*B, H) bf16 workspace
    float* __restrict__ h_final)       // (B, H)
{
    __shared__ unsigned short Whh_s[HID * LDK]; // [j][k] = bf16(W_hh[j][k])
    __shared__ unsigned short Hs[16 * LDK];     // current hidden state tile (bf16)

    const int tid  = threadIdx.x;
    const int lane = tid & 31;
    const int wave = tid >> 5;           // 0..7
    const int b0   = blockIdx.x * 16;    // batch offset

    // --- TDM: DMA W_hh (bf16) into padded LDS; one wave issues ---
    if (wave == 0) {
        tdm_load_tile_bf16((unsigned int)(uintptr_t)(void*)&Whh_s[0], Whh_bf, HID, HID);
        __builtin_amdgcn_s_wait_tensorcnt(0);
    }
    // --- init hidden state from `state` (16 x 256) ---
#pragma unroll
    for (int c = 0; c < 16; ++c) {
        Hs[c * LDK + tid] = (unsigned short)cvt_bf16(state[(b0 + c) * HID + tid]);
    }
    __syncthreads();

    const int n     = lane & 15;
    const int mbase = (lane >> 4) << 3;  // 0 or 8
    float bias0 = b_ih[wave * 32 +      n] + b_hh[wave * 32 +      n];
    float bias1 = b_ih[wave * 32 + 16 + n] + b_hh[wave * 32 + 16 + n];

    for (int t = 0; t < SEQ; ++t) {
        v8f acc0 = {0.f,0.f,0.f,0.f,0.f,0.f,0.f,0.f};
        v8f acc1 = {0.f,0.f,0.f,0.f,0.f,0.f,0.f,0.f};
#pragma unroll
        for (int ks = 0; ks < HID; ks += 32) {
            v16bf a   = load_frag_a(Hs, n, ks, lane);
            v16bf bb0 = load_frag_b(Whh_s, wave * 32 +      n, ks, lane);
            v16bf bb1 = load_frag_b(Whh_s, wave * 32 + 16 + n, ks, lane);
            acc0 = __builtin_amdgcn_wmma_f32_16x16x32_bf16(false, a, false, bb0, (short)0, acc0, false, false);
            acc1 = __builtin_amdgcn_wmma_f32_16x16x32_bf16(false, a, false, bb1, (short)0, acc1, false, false);
        }
        __syncthreads();   // all waves done reading Hs for step t

#pragma unroll
        for (int tn = 0; tn < 2; ++tn) {
            FragC ac; ac.v = tn ? acc1 : acc0;
            int   ng = wave * 32 + tn * 16 + n;     // hidden index (column)
            float bs = tn ? bias1 : bias0;
#pragma unroll
            for (int r = 0; r < 8; ++r) {
                int m   = r + mbase;                 // local batch row
                int b   = b0 + m;
                int idx = X[b * SEQ + t];
                float xp  = W_ih[(size_t)ng * VOCAB + idx];
                float val = tanhf(ac.f[r] + xp + bs);
                unsigned short hb = (unsigned short)cvt_bf16(val);
                Hs[m * LDK + ng] = hb;
                Ybf[((size_t)(t * BATCH + b)) * HID + ng] = hb;
                if (t == SEQ - 1) h_final[b * HID + ng] = val;
            }
        }
        __syncthreads();   // new Hs visible before next step
    }
}

// ---------------------------------------------------------------------------
// Stage 3: out = Y @ W_out^T + b_out.  (16384 x 256) @ (256 x 8192).
// 128x128 tile per block, K=256 TDM-staged in LDS. 8 waves = 4(m) x 2(n),
// each wave computes 32x64 = 2x4 tiles of 16x16. Non-temporal f32 stores.
// ---------------------------------------------------------------------------
__global__ __launch_bounds__(256) void out_proj_kernel(
    const unsigned short* __restrict__ Ybf,      // (T*B, H) bf16
    const unsigned short* __restrict__ Wout_bf,  // (V, H)  bf16
    const float* __restrict__ b_out,             // (V)
    float* __restrict__ out)                     // (T*B, V)
{
    __shared__ unsigned short Ys[128 * LDK];
    __shared__ unsigned short Ws[128 * LDK];

    const int tid  = threadIdx.x;
    const int lane = tid & 31;
    const int wave = tid >> 5;
    const int wm   = wave >> 1;          // 0..3
    const int wn   = wave & 1;           // 0..1
    const int r0   = blockIdx.x * 128;   // output row block
    const int v0   = blockIdx.y * 128;   // output col (vocab) block

    // --- TDM: DMA both tiles into padded LDS; one wave issues ---
    if (wave == 0) {
        tdm_load_tile_bf16((unsigned int)(uintptr_t)(void*)&Ys[0],
                           Ybf + (size_t)r0 * HID, 128, SEQ * BATCH);
        tdm_load_tile_bf16((unsigned int)(uintptr_t)(void*)&Ws[0],
                           Wout_bf + (size_t)v0 * HID, 128, VOCAB);
        __builtin_amdgcn_s_wait_tensorcnt(0);
    }
    __syncthreads();

    v8f acc[2][4];
#pragma unroll
    for (int tm = 0; tm < 2; ++tm)
#pragma unroll
        for (int tn = 0; tn < 4; ++tn)
            acc[tm][tn] = (v8f){0.f,0.f,0.f,0.f,0.f,0.f,0.f,0.f};

    const int n = lane & 15;
#pragma unroll
    for (int ks = 0; ks < HID; ks += 32) {
        v16bf a[2], bb[4];
#pragma unroll
        for (int tm = 0; tm < 2; ++tm)
            a[tm] = load_frag_a(Ys, wm * 32 + tm * 16 + n, ks, lane);
#pragma unroll
        for (int tn = 0; tn < 4; ++tn)
            bb[tn] = load_frag_b(Ws, wn * 64 + tn * 16 + n, ks, lane);
#pragma unroll
        for (int tm = 0; tm < 2; ++tm)
#pragma unroll
            for (int tn = 0; tn < 4; ++tn)
                acc[tm][tn] = __builtin_amdgcn_wmma_f32_16x16x32_bf16(
                    false, a[tm], false, bb[tn], (short)0, acc[tm][tn], false, false);
    }

    const int mbase = (lane >> 4) << 3;
#pragma unroll
    for (int tn = 0; tn < 4; ++tn) {
        int   col = v0 + wn * 64 + tn * 16 + n;
        float bo  = b_out[col];
#pragma unroll
        for (int tm = 0; tm < 2; ++tm) {
            FragC ac; ac.v = acc[tm][tn];
#pragma unroll
            for (int r = 0; r < 8; ++r) {
                int row = r0 + wm * 32 + tm * 16 + r + mbase;
                __builtin_nontemporal_store(ac.f[r] + bo, &out[(size_t)row * VOCAB + col]);
            }
        }
    }
}

extern "C" void kernel_launch(void* const* d_in, const int* in_sizes, int n_in,
                              void* d_out, int out_size, void* d_ws, size_t ws_size,
                              hipStream_t stream) {
    const int*   X     = (const int*)  d_in[0];
    const float* state = (const float*)d_in[1];
    const float* W_ih  = (const float*)d_in[2];
    const float* W_hh  = (const float*)d_in[3];
    const float* b_ih  = (const float*)d_in[4];
    const float* b_hh  = (const float*)d_in[5];
    const float* W_out = (const float*)d_in[6];
    const float* b_out = (const float*)d_in[7];

    float* out     = (float*)d_out;
    float* h_final = out + (size_t)SEQ * BATCH * VOCAB;       // second tuple element

    // workspace layout (bf16): Y (8MB) | W_out (4MB) | W_hh (128KB)
    unsigned short* Ybf     = (unsigned short*)d_ws;
    unsigned short* Wout_bf = Ybf + (size_t)SEQ * BATCH * HID;
    unsigned short* Whh_bf  = Wout_bf + (size_t)VOCAB * HID;

    int conv_chunks = WOUT_CHUNKS + WHH_CHUNKS;
    convert_weights_kernel<<<(conv_chunks + 255) / 256, 256, 0, stream>>>(W_out, W_hh, Wout_bf, Whh_bf);
    rnn_scan_kernel<<<4, 256, 0, stream>>>(X, state, W_ih, Whh_bf, b_ih, b_hh, Ybf, h_final);
    out_proj_kernel<<<dim3(128, 64), 256, 0, stream>>>(Ybf, Wout_bf, b_out, out);
}
